// SOARALinear_4827543241364
// MI455X (gfx1250) — compile-verified, matrix operand order checked by reading, not run on hardware
//
#include <hip/hip_runtime.h>
#include <hip/hip_bf16.h>

// ---- WMMA vector types (gfx1250, wave32) ----
typedef __attribute__((ext_vector_type(16))) __bf16 v16bf;
typedef __attribute__((ext_vector_type(8)))  __bf16 v8bf;
typedef __attribute__((ext_vector_type(8)))  float  v8f;

#define BM 128               // block tile rows (A / C rows)
#define BN 128               // block tile cols (B rows = C cols)
#define BK 32                // K-chunk = WMMA bf16 K
#define LDS_STRIDE 40        // bf16 per LDS row: 32 data + 8 pad (80B, 16B-aligned)

// C[N,M] = A[N,K] @ B[M,K]^T  (+ bias[M]) (+= into C)
// Requires N%128==0, M%128==0, K%32==0 (true for all calls here).
// 256 threads = 8 waves laid out 4(m) x 2(n); wave tile 32x64 = 2x4 WMMA tiles.
// Double-buffered LDS: prefetch chunk k+1 from global into registers while
// WMMAs consume chunk k; one barrier per K-chunk.
template<bool ADD_BIAS, bool ACCUM>
__global__ __launch_bounds__(256)
void gemm_bt_wmma(const float* __restrict__ A, const float* __restrict__ B,
                  const float* __restrict__ bias, float* __restrict__ C,
                  int N, int M, int K)
{
    __shared__ __bf16 As[2][BM * LDS_STRIDE];
    __shared__ __bf16 Bs[2][BN * LDS_STRIDE];

    const int tid  = threadIdx.x;
    const int lane = tid & 31;
    const int wave = tid >> 5;
    const int wm   = wave >> 1;        // 0..3
    const int wn   = wave & 1;         // 0..1

    const int rowBase = blockIdx.y * BM;
    const int colBase = blockIdx.x * BN;

    // Per-thread staging coordinates: 1024 float4 per 128x32 tile, 4 per thread.
    // id = tid + i*256 ; r = id>>3 ; c4 = (id&7)*4
    const int r0  = tid >> 3;            // rows r0, r0+32, r0+64, r0+96
    const int c4  = (tid & 7) << 2;

    const float* Ap = A + (size_t)(rowBase + r0) * K + c4;
    const float* Bp = B + (size_t)(colBase + r0) * K + c4;
    const size_t rowStep = (size_t)32 * K;   // 32 rows between staging slices

    v8f acc[2][4];
#pragma unroll
    for (int i = 0; i < 2; ++i)
#pragma unroll
        for (int j = 0; j < 4; ++j) acc[i][j] = v8f{};

    // ISA 16-bit A/B fragment layout: lane<16 -> K 0..7 & 16..23 of row (lane&15);
    // lane>=16 -> K 8..15 & 24..31.  Two 16B LDS loads per fragment.
    const int halfsel = lane >> 4;         // 0 or 1
    const int kb      = halfsel * 8;       // bf16 offset within K-chunk
    const int fr      = lane & 15;         // fragment row (M or N within 16-tile)

    float4 ra[4], rb[4];

    // ---- prologue: fetch chunk 0 and stage into buffer 0 ----
#pragma unroll
    for (int i = 0; i < 4; ++i) {
        ra[i] = *(const float4*)(Ap + i * rowStep);
        rb[i] = *(const float4*)(Bp + i * rowStep);
    }
#pragma unroll
    for (int i = 0; i < 4; ++i) {
        __bf16* da = &As[0][(r0 + 32 * i) * LDS_STRIDE + c4];
        da[0] = (__bf16)ra[i].x; da[1] = (__bf16)ra[i].y;
        da[2] = (__bf16)ra[i].z; da[3] = (__bf16)ra[i].w;
        __bf16* db = &Bs[0][(r0 + 32 * i) * LDS_STRIDE + c4];
        db[0] = (__bf16)rb[i].x; db[1] = (__bf16)rb[i].y;
        db[2] = (__bf16)rb[i].z; db[3] = (__bf16)rb[i].w;
    }
    __syncthreads();

    const int nk = K / BK;
    for (int kc = 0; kc < nk; ++kc) {
        const int cur = kc & 1;
        const int nxt = cur ^ 1;
        const bool more = (kc + 1) < nk;

        // issue next chunk's global loads early (overlaps with LDS reads + WMMA)
        if (more) {
            const float* Ak = Ap + (size_t)(kc + 1) * BK;
            const float* Bk = Bp + (size_t)(kc + 1) * BK;
#pragma unroll
            for (int i = 0; i < 4; ++i) {
                ra[i] = *(const float4*)(Ak + i * rowStep);
                rb[i] = *(const float4*)(Bk + i * rowStep);
            }
        }

        // fragment loads from current buffer
        v16bf afrag[2], bfrag[4];
#pragma unroll
        for (int i = 0; i < 2; ++i) {
            const int ar = (wm * 2 + i) * 16 + fr;
            const __bf16* p = &As[cur][ar * LDS_STRIDE + kb];
            v8bf lo = *(const v8bf*)p;
            v8bf hi = *(const v8bf*)(p + 16);
            afrag[i] = __builtin_shufflevector(lo, hi, 0,1,2,3,4,5,6,7,8,9,10,11,12,13,14,15);
        }
#pragma unroll
        for (int j = 0; j < 4; ++j) {
            const int br = (wn * 4 + j) * 16 + fr;
            const __bf16* p = &Bs[cur][br * LDS_STRIDE + kb];
            v8bf lo = *(const v8bf*)p;
            v8bf hi = *(const v8bf*)(p + 16);
            bfrag[j] = __builtin_shufflevector(lo, hi, 0,1,2,3,4,5,6,7,8,9,10,11,12,13,14,15);
        }

        // stage next chunk into the other buffer (no conflict with reads above)
        if (more) {
#pragma unroll
            for (int i = 0; i < 4; ++i) {
                __bf16* da = &As[nxt][(r0 + 32 * i) * LDS_STRIDE + c4];
                da[0] = (__bf16)ra[i].x; da[1] = (__bf16)ra[i].y;
                da[2] = (__bf16)ra[i].z; da[3] = (__bf16)ra[i].w;
                __bf16* db = &Bs[nxt][(r0 + 32 * i) * LDS_STRIDE + c4];
                db[0] = (__bf16)rb[i].x; db[1] = (__bf16)rb[i].y;
                db[2] = (__bf16)rb[i].z; db[3] = (__bf16)rb[i].w;
            }
        }
        __syncthreads();

#pragma unroll
        for (int i = 0; i < 2; ++i)
#pragma unroll
            for (int j = 0; j < 4; ++j)
                acc[i][j] = __builtin_amdgcn_wmma_f32_16x16x32_bf16(
                    false, afrag[i], false, bfrag[j],
                    (short)0, acc[i][j], false, false);
    }

    // Epilogue. C/D layout: VGPR v, lanes 0-15 -> M=v, N=lane; lanes 16-31 -> M=v+8.
#pragma unroll
    for (int i = 0; i < 2; ++i) {
#pragma unroll
        for (int j = 0; j < 4; ++j) {
            const int cg = colBase + (wn * 4 + j) * 16 + fr;
            float bb;
            if (ADD_BIAS) bb = bias[cg];
#pragma unroll
            for (int v = 0; v < 8; ++v) {
                const int rg = rowBase + (wm * 2 + i) * 16 + v + halfsel * 8;
                const size_t off = (size_t)rg * M + cg;
                float val = acc[i][j][v];
                if (ADD_BIAS) val += bb;
                if (ACCUM) C[off] += val; else C[off] = val;
            }
        }
    }
}

// Precompute (cos, sin) tables for both butterfly parameter sets.
__global__ void build_tables(const float* __restrict__ thu, const float* __restrict__ thv,
                             float2* __restrict__ cs_u, float2* __restrict__ cs_v, int n)
{
    const int i = blockIdx.x * blockDim.x + threadIdx.x;
    if (i < n) {
        const float tu = thu[i];
        const float tv = thv[i];
        cs_u[i] = make_float2(cosf(tu), sinf(tu));
        cs_v[i] = make_float2(cosf(tv), sinf(tv));
    }
}

// Apply h <- ((h @ R_V) * S) @ R_U^T row-wise, R fixed at 1024 (m=10 stages).
// R_V = C0 C1 ... C9 (apply C0 first); R_U^T = C9^T ... C0^T (apply C9^T first).
// Level j: half = 512>>j; pair u: b=u>>(9-j), t=u&(half-1), p=b*2*half+t, q=p+half.
__global__ __launch_bounds__(256)
void butterfly_mid(float* __restrict__ h, const float2* __restrict__ cs_v,
                   const float2* __restrict__ cs_u, const float* __restrict__ S)
{
    __shared__ float row[1024];
    const int tid = threadIdx.x;
    float* hr = h + (size_t)blockIdx.x * 1024;

#pragma unroll
    for (int i = 0; i < 4; ++i) row[tid + i * 256] = hr[tid + i * 256];
    __syncthreads();

    // forward stages of R_V: out[p] = hp*c + hq*s ; out[q] = -hp*s + hq*c
#pragma unroll
    for (int j = 0; j < 10; ++j) {
        const int shift = 9 - j;                 // log2(half)
#pragma unroll
        for (int t = 0; t < 2; ++t) {
            const int u   = tid + t * 256;       // pair index 0..511
            const int b   = u >> shift;
            const int off = u & ((1 << shift) - 1);
            const int p   = (b << (shift + 1)) + off;
            const int q   = p + (1 << shift);
            const float2 cs = cs_v[j * 512 + u];
            const float hp = row[p], hq = row[q];
            row[p] =  hp * cs.x + hq * cs.y;
            row[q] = -hp * cs.y + hq * cs.x;
        }
        __syncthreads();
    }

#pragma unroll
    for (int i = 0; i < 4; ++i) row[tid + i * 256] *= S[tid + i * 256];
    __syncthreads();

    // transposed stages of R_U: out[p] = hp*c - hq*s ; out[q] = hp*s + hq*c
#pragma unroll
    for (int j = 9; j >= 0; --j) {
        const int shift = 9 - j;
#pragma unroll
        for (int t = 0; t < 2; ++t) {
            const int u   = tid + t * 256;
            const int b   = u >> shift;
            const int off = u & ((1 << shift) - 1);
            const int p   = (b << (shift + 1)) + off;
            const int q   = p + (1 << shift);
            const float2 cs = cs_u[j * 512 + u];
            const float hp = row[p], hq = row[q];
            row[p] = hp * cs.x - hq * cs.y;
            row[q] = hp * cs.y + hq * cs.x;
        }
        __syncthreads();
    }

#pragma unroll
    for (int i = 0; i < 4; ++i) hr[tid + i * 256] = row[tid + i * 256];
}

extern "C" void kernel_launch(void* const* d_in, const int* in_sizes, int n_in,
                              void* d_out, int out_size, void* d_ws, size_t ws_size,
                              hipStream_t stream)
{
    const float* x    = (const float*)d_in[0];   // [N, DIN]
    const float* W    = (const float*)d_in[1];   // [DOUT, DIN]
    const float* bias = (const float*)d_in[2];   // [DOUT]
    const float* U    = (const float*)d_in[3];   // [DOUT, R]
    const float* S    = (const float*)d_in[4];   // [R]
    const float* V    = (const float*)d_in[5];   // [R, DIN]
    const float* thu  = (const float*)d_in[6];   // [10, R/2]
    const float* thv  = (const float*)d_in[7];   // [10, R/2]

    const int R    = in_sizes[4];                // 1024
    const int DOUT = in_sizes[2];                // 4096
    const int DIN  = in_sizes[5] / R;            // 4096
    const int N    = in_sizes[0] / DIN;          // 16384

    float* out = (float*)d_out;
    char*  ws  = (char*)d_ws;
    float2* cs_u = (float2*)ws;                  // 10*512 float2 = 40 KB
    float2* cs_v = (float2*)(ws + (64 << 10));   // 40 KB
    float*  h    = (float*)(ws + (1 << 20));     // N*R fp32 = 64 MB

    // 0) sin/cos tables for both butterfly stacks
    const int nth = 10 * (R / 2);
    build_tables<<<(nth + 255) / 256, 256, 0, stream>>>(thu, thv, cs_u, cs_v, nth);

    // 1) h = x @ V^T                      [N, R]
    gemm_bt_wmma<false, false><<<dim3(R / BN, N / BM), 256, 0, stream>>>(
        x, V, nullptr, h, N, R, DIN);

    // 2) h = ((h @ R_V) * S) @ R_U^T     (butterfly stages applied directly)
    butterfly_mid<<<N, 256, 0, stream>>>(h, cs_v, cs_u, S);

    // 3) out = x @ W^T + bias            [N, DOUT]
    gemm_bt_wmma<true, false><<<dim3(DOUT / BN, N / BM), 256, 0, stream>>>(
        x, W, bias, out, N, DOUT, DIN);

    // 4) out += h @ U^T
    gemm_bt_wmma<false, true><<<dim3(DOUT / BN, N / BM), 256, 0, stream>>>(
        h, U, nullptr, out, N, DOUT, R);
}
